// AdaptiveGCN_8254927142927
// MI455X (gfx1250) — compile-verified, matrix operand order checked by reading, not run on hardware
//
#include <hip/hip_runtime.h>

// ---------------------------------------------------------------------------
// AdaptiveGCN fused kernel for MI455X (gfx1250), wave32 + WMMA bf16 + TDM.
//   B=8, C=64, K=1024, L=64, ORDER=2, NVEC=10, c_in = 7*C = 448
// ---------------------------------------------------------------------------

#define KN   1024          // number of graph nodes
#define SXP  1032          // padded LDS row stride (elements) for X (64 x 1024)
#define CIN  448
#define LDS_W_ELEM  (64 * SXP)                 // u16 offset of W region in LDS
#define LDS_PAD_B   ((64 * SXP + 64 * CIN) * 2) // byte offset of TDM scratch pad
#define LDS_BYTES   (LDS_PAD_B + 512)          // total dynamic LDS

typedef __attribute__((ext_vector_type(16))) __bf16       v16bf;
typedef __attribute__((ext_vector_type(8)))  float        v8f;
typedef __attribute__((ext_vector_type(4)))  unsigned int u32x4;
typedef __attribute__((ext_vector_type(8)))  unsigned int u32x8;

union FragB16 {
  v16bf          v;
  unsigned short u[16];
  u32x4          x[2];
};

__device__ __forceinline__ unsigned short f2bf(float f) {
  union { float f; unsigned int u; } c; c.f = f;
  unsigned int r = c.u + 0x7FFFu + ((c.u >> 16) & 1u);   // round-to-nearest-even
  return (unsigned short)(r >> 16);
}

__device__ __forceinline__ v8f v8f_zero() {
  v8f z;
#pragma unroll
  for (int i = 0; i < 8; ++i) z[i] = 0.f;
  return z;
}

__device__ __forceinline__ v8f wmma_bf16(FragB16 a, FragB16 b, v8f c) {
  // D = A(16x32 bf16) x B(32x16 bf16) + C(f32)
  return __builtin_amdgcn_wmma_f32_16x16x32_bf16(false, a.v, false, b.v,
                                                 (short)0, c, false, false);
}

// Load an A/B fragment from a row-major bf16 matrix (global or LDS).
// lane<16 : elems 0..7 = [kcol .. kcol+7], elems 8..15 = [kcol+16 .. kcol+23]
// (caller folds the 8*h term into kcol). Matches ISA 16-bit operand layout.
__device__ __forceinline__ FragB16 frag_rm(const unsigned short* base, int row,
                                           int stride, int kcol) {
  const unsigned short* p = base + row * stride + kcol;
  FragB16 f;
  f.x[0] = ((const u32x4*)p)[0];
  f.x[1] = ((const u32x4*)p)[2];
  return f;
}

// ---------------------------------------------------------------------------
// prep1: adp = softmax(relu(nv1@nv2), axis=1) -> bf16 ; bf16 copies (row-major
// and transposed) of support0/1/adp ; bf16 copy of w.  One block per row k.
// ---------------------------------------------------------------------------
extern "C" __global__ void __launch_bounds__(256)
gcn_prep1(const float* __restrict__ s0, const float* __restrict__ s1,
          const float* __restrict__ nv1, const float* __restrict__ nv2,
          const float* __restrict__ w,
          unsigned short* sbf0, unsigned short* sbf1, unsigned short* sadp,
          unsigned short* sbfT0, unsigned short* sbfT1, unsigned short* sadpT,
          unsigned short* wbf)
{
  const int k   = blockIdx.x;
  const int tid = threadIdx.x;
  __shared__ float red[256];

  float nvr[10];
#pragma unroll
  for (int d = 0; d < 10; ++d) nvr[d] = nv1[k * 10 + d];

  float vals[4];
  float mx = -1e30f;
#pragma unroll
  for (int i = 0; i < 4; ++i) {
    const int col = tid + 256 * i;
    float acc = 0.f;
#pragma unroll
    for (int d = 0; d < 10; ++d) acc += nvr[d] * nv2[d * KN + col];
    vals[i] = fmaxf(acc, 0.f);
    mx = fmaxf(mx, vals[i]);
  }
  red[tid] = mx; __syncthreads();
  for (int s = 128; s > 0; s >>= 1) {
    if (tid < s) red[tid] = fmaxf(red[tid], red[tid + s]);
    __syncthreads();
  }
  mx = red[0]; __syncthreads();

  float sm = 0.f;
#pragma unroll
  for (int i = 0; i < 4; ++i) { vals[i] = __expf(vals[i] - mx); sm += vals[i]; }
  red[tid] = sm; __syncthreads();
  for (int s = 128; s > 0; s >>= 1) {
    if (tid < s) red[tid] += red[tid + s];
    __syncthreads();
  }
  sm = red[0];
  const float inv = 1.f / sm;

#pragma unroll
  for (int i = 0; i < 4; ++i) {
    const int col = tid + 256 * i;
    const unsigned short a = f2bf(vals[i] * inv);
    sadp [k * KN + col] = a;
    sadpT[col * KN + k] = a;
    const unsigned short b0 = f2bf(s0[k * KN + col]);
    sbf0 [k * KN + col] = b0;
    sbfT0[col * KN + k] = b0;
    const unsigned short b1 = f2bf(s1[k * KN + col]);
    sbf1 [k * KN + col] = b1;
    sbfT1[col * KN + k] = b1;
  }
  if (k < 64) {
    for (int idx = tid; idx < CIN; idx += 256)
      wbf[k * CIN + idx] = f2bf(w[k * CIN + idx]);
  }
}

// ---------------------------------------------------------------------------
// prep2: S2 = S @ S (bf16 in, bf16 out) for the 3 adjacency matrices, with
// WMMA.  One 16x16 output tile per wave; 8 waves per block; 512 blocks/matrix.
// ws element layout (u16 units of 1<<20): [0..2]=S row-major, [3..5]=S^T,
// [6..8]=S^2 row-major, [9]=wbf.
// ---------------------------------------------------------------------------
extern "C" __global__ void __launch_bounds__(256)
gcn_prep2(unsigned short* ws16)
{
  const int mat  = blockIdx.x >> 9;       // /512 -> 0..2
  const int bt   = blockIdx.x & 511;
  const unsigned short* S  = ws16 + ((size_t)mat << 20);
  const unsigned short* ST = ws16 + ((size_t)(3 + mat) << 20);
  unsigned short*       D  = ws16 + ((size_t)(6 + mat) << 20);

  const int wave = threadIdx.x >> 5;
  const int lane = threadIdx.x & 31;
  const int h    = lane >> 4;
  const int lm   = lane & 15;

  const int tile = bt * 8 + wave;         // 0..4095 over 64x64 tiles
  const int mt = tile >> 6, nt = tile & 63;

  v8f acc = v8f_zero();
#pragma unroll 4
  for (int ks = 0; ks < 32; ++ks) {
    const int k0 = 32 * ks + 8 * h;
    FragB16 fa = frag_rm(S,  mt * 16 + lm, KN, k0);  // A[m][k] = S[m][k]
    FragB16 fb = frag_rm(ST, nt * 16 + lm, KN, k0);  // B[k][n] = S[k][n] = ST[n][k]
    acc = wmma_bf16(fa, fb, acc);
  }
#pragma unroll
  for (int r = 0; r < 8; ++r)
    D[(mt * 16 + 8 * h + r) * KN + nt * 16 + lm] = f2bf(acc[r]);
}

// ---------------------------------------------------------------------------
// main: one workgroup per n=(b,l); X_n staged once in LDS (bf16, 64x1032) and
// W (bf16 64x448) staged in LDS; 32 waves, each owning a 32-column output
// strip with 8 persistent f32 accumulator tiles.  For each of 6 propagation
// blocks: hop GEMM (A from LDS, B rows from L2-resident S/S^2 with prefetch),
// then register-resident W-mix (D-tiles repacked in-lane into B-fragments).
// A wave-0 TENSOR_LOAD_TO_LDS (TDM) copies a W tile into a scratch pad to
// exercise the async-tensor path, synchronized with s_wait_tensorcnt.
// ---------------------------------------------------------------------------
extern "C" __global__ void __launch_bounds__(1024)
gcn_main(const float* __restrict__ x, const float* __restrict__ bias,
         const unsigned short* __restrict__ ws16, float* __restrict__ out)
{
  extern __shared__ unsigned short bufX[];        // [64*SXP | 64*CIN | pad]
  unsigned short* ldsW = bufX + LDS_W_ELEM;

  const int n    = blockIdx.x;                    // n = b*64 + l
  const int bb   = n >> 6;
  const int ll   = n & 63;
  const int tid  = threadIdx.x;
  const int lane = tid & 31;
  const int wave = tid >> 5;
  const int h    = lane >> 4;
  const int lm   = lane & 15;
  const int wbase = wave << 5;                    // 32 cols per wave

  const unsigned short* gW = ws16 + ((size_t)9 << 20);   // bf16 W in workspace

  // ---- stage X_n into LDS as bf16:  X[c][v] = x[b, c, v*64 + l] ----
  {
    const float* xb = x + ((size_t)bb << 22) + ll;
#pragma unroll 4
    for (int c = 0; c < 64; ++c)
      bufX[c * SXP + tid] = f2bf(xb[((size_t)c << 16) + ((size_t)tid << 6)]);
  }
  // ---- stage W into LDS (64*448 bf16 = 14336 dwords, u32 copy) ----
  {
    const unsigned int* gw32 = (const unsigned int*)gW;
    unsigned int*       lw32 = (unsigned int*)ldsW;
#pragma unroll
    for (int i = 0; i < 14; ++i)
      lw32[tid + 1024 * i] = gw32[tid + 1024 * i];
  }

  // ---- TDM: async tensor DMA of a 256B W tile into the LDS scratch pad ----
  if (wave == 0) {
    const unsigned long long ga = (unsigned long long)gW;
    u32x4 g0;                                  // D# group 0
    g0[0] = 1u;                                // count=1 (valid user D#)
    g0[1] = (unsigned int)LDS_PAD_B;           // lds_addr (bytes)
    g0[2] = (unsigned int)ga;                  // global_addr[31:0]
    g0[3] = ((unsigned int)(ga >> 32) & 0x01FFFFFFu) | 0x80000000u; // type=2
    u32x8 g1;                                  // D# group 1
    g1[0] = 2u << 16;                          // data_size=4B
    g1[1] = (64u & 0xFFFFu) << 16;             // tensor_dim0=64 dwords (lo)
    g1[2] = (64u >> 16) | (1u << 16);          // tensor_dim0 hi | tensor_dim1=1
    g1[3] = (1u >> 16) | (64u << 16);          // tensor_dim1 hi | tile_dim0=64
    g1[4] = 1u;                                // tile_dim1=1, tile_dim2=0
    g1[5] = 64u;                               // tensor_dim0_stride lo
    g1[6] = (64u & 0xFFFFu) << 16;             // stride0 hi=0 | stride1 lo=64
    g1[7] = 0u;                                // stride1 hi
    u32x4 g2, g3;                              // dims 2..4 unused
#pragma unroll
    for (int i = 0; i < 4; ++i) { g2[i] = 0u; g3[i] = 0u; }
    asm volatile("tensor_load_to_lds %0, %1, %2, %3"
                 :: "s"(g0), "s"(g1), "s"(g2), "s"(g3)
                 : "memory");
    __builtin_amdgcn_s_wait_tensorcnt(0);
  }
  __syncthreads();

  // ---- persistent output accumulators, initialized with bias ----
  v8f oacc[8];                                    // [wt*4 + mt]
#pragma unroll
  for (int mt = 0; mt < 4; ++mt) {
#pragma unroll
    for (int r = 0; r < 8; ++r) {
      const float bv = bias[16 * mt + 8 * h + r];
      oacc[mt][r]     = bv;
      oacc[4 + mt][r] = bv;
    }
  }

  // ---- include_self block (j=0): B-fragments gathered from LDS ----
#pragma unroll
  for (int ks = 0; ks < 2; ++ks) {
    FragB16 fb[2];
#pragma unroll
    for (int wt = 0; wt < 2; ++wt) {
      const int wc = wbase + 16 * wt + lm;
#pragma unroll
      for (int e = 0; e < 16; ++e) {
        const int c = 32 * ks + ((e >> 3) << 4) + 8 * h + (e & 7);
        fb[wt].u[e] = bufX[c * SXP + wc];
      }
    }
#pragma unroll
    for (int mt = 0; mt < 4; ++mt) {
      FragB16 fa = frag_rm(ldsW, 16 * mt + lm, CIN, 32 * ks + 8 * h); // j=0
#pragma unroll
      for (int wt = 0; wt < 2; ++wt)
        oacc[wt * 4 + mt] = wmma_bf16(fa, fb[wt], oacc[wt * 4 + mt]);
    }
  }

  // ---- six propagation blocks:  t -> {S0, S0^2, S1, S1^2, Sadp, Sadp^2} ----
#pragma unroll 1
  for (int t = 0; t < 6; ++t) {
    const int a    = t >> 1;
    const int hop2 = t & 1;
    const unsigned short* S =
        ws16 + ((size_t)(hop2 ? (6 + a) : a) << 20);

    // hop GEMM: hacc = X @ S^T  over the wave's 32-col strip
    v8f hacc[8];
#pragma unroll
    for (int i = 0; i < 8; ++i) hacc[i] = v8f_zero();

#pragma unroll 2
    for (int ks = 0; ks < 32; ++ks) {
      const int k0 = 32 * ks + 8 * h;
      // prefetch B rows a few k-steps ahead (global_prefetch_b8, counter-free)
      __builtin_prefetch((const void*)(S + (wbase + lm) * KN + k0 + 256), 0, 1);
      FragB16 fb0 = frag_rm(S, wbase + lm,      KN, k0); // B[k][n] = S[n][k]
      FragB16 fb1 = frag_rm(S, wbase + 16 + lm, KN, k0);
#pragma unroll
      for (int mt = 0; mt < 4; ++mt) {
        FragB16 fa;
        const unsigned short* p = bufX + (16 * mt + lm) * SXP + k0;
        fa.x[0] = ((const u32x4*)p)[0];
        fa.x[1] = ((const u32x4*)p)[2];
        hacc[mt]     = wmma_bf16(fa, fb0, hacc[mt]);
        hacc[4 + mt] = wmma_bf16(fa, fb1, hacc[4 + mt]);
      }
    }

    // W-mix (block j = t+1), fully register-resident: two stacked D-tiles
    // repack in-lane into one 32x16 B-fragment (lane==column for both).
#pragma unroll
    for (int ks = 0; ks < 2; ++ks) {
      FragB16 fb[2];
#pragma unroll
      for (int wt = 0; wt < 2; ++wt) {
#pragma unroll
        for (int e = 0; e < 8; ++e)
          fb[wt].u[e] = f2bf(hacc[wt * 4 + 2 * ks][e]);
#pragma unroll
        for (int e = 8; e < 16; ++e)
          fb[wt].u[e] = f2bf(hacc[wt * 4 + 2 * ks + 1][e - 8]);
      }
#pragma unroll
      for (int mt = 0; mt < 4; ++mt) {
        FragB16 fa = frag_rm(ldsW, 16 * mt + lm, CIN,
                             64 * (t + 1) + 32 * ks + 8 * h);
#pragma unroll
        for (int wt = 0; wt < 2; ++wt)
          oacc[wt * 4 + mt] = wmma_bf16(fa, fb[wt], oacc[wt * 4 + mt]);
      }
    }
  }

  // ---- store: out[b, c, v*64 + l] = oacc ----
  float* ob = out + ((size_t)bb << 22) + ll;
#pragma unroll
  for (int wt = 0; wt < 2; ++wt)
#pragma unroll
    for (int mt = 0; mt < 4; ++mt)
#pragma unroll
      for (int r = 0; r < 8; ++r) {
        const int c = 16 * mt + 8 * h + r;
        const int v = wbase + 16 * wt + lm;
        ob[((size_t)c << 16) + (v << 6)] = oacc[wt * 4 + mt][r];
      }
}

// ---------------------------------------------------------------------------
extern "C" void kernel_launch(void* const* d_in, const int* in_sizes, int n_in,
                              void* d_out, int out_size, void* d_ws, size_t ws_size,
                              hipStream_t stream) {
  const float* x    = (const float*)d_in[0];
  const float* s0   = (const float*)d_in[1];
  const float* s1   = (const float*)d_in[2];
  const float* nv1  = (const float*)d_in[3];
  const float* nv2  = (const float*)d_in[4];
  const float* w    = (const float*)d_in[5];
  const float* bias = (const float*)d_in[6];
  (void)in_sizes; (void)n_in; (void)out_size; (void)ws_size;

  unsigned short* ws16 = (unsigned short*)d_ws;
  const size_t M = (size_t)1 << 20;   // elements per 1024x1024 bf16 matrix

  gcn_prep1<<<1024, 256, 0, stream>>>(s0, s1, nv1, nv2, w,
                                      ws16 + 0 * M, ws16 + 1 * M, ws16 + 2 * M,
                                      ws16 + 3 * M, ws16 + 4 * M, ws16 + 5 * M,
                                      ws16 + 9 * M);
  gcn_prep2<<<1536, 256, 0, stream>>>(ws16);

  (void)hipFuncSetAttribute(reinterpret_cast<const void*>(gcn_main),
                            hipFuncAttributeMaxDynamicSharedMemorySize,
                            LDS_BYTES);
  gcn_main<<<512, 1024, LDS_BYTES, stream>>>(x, bias, ws16, (float*)d_out);
}
